// Seq2SeqAttn_52518860095509
// MI455X (gfx1250) — matrix operand order repaired and checked by reading, not compiled, not run on hardware
//
#include <hip/hip_runtime.h>
#include <hip/hip_bf16.h>
#include <math.h>

typedef __attribute__((ext_vector_type(16))) __bf16 v16bf;
typedef __attribute__((ext_vector_type(8)))  __bf16 v8bf;
typedef __attribute__((ext_vector_type(4)))  __bf16 v4bf;
typedef __attribute__((ext_vector_type(4)))  float  v4f;
typedef __attribute__((ext_vector_type(8)))  float  v8f;

namespace {
constexpr int kB   = 32;
constexpr int kCL  = 400;
constexpr int kQL  = 30;
constexpr int kEMB = 300;
constexpr int kH   = 600;
constexpr int kH2  = 1200;
constexpr int kH4  = 2400;
constexpr int kV   = 50000;
}

// ---------------------------------------------------------------------------
// Weight prep: out (N32 x K32, bf16, zero-padded) = transpose of in[K x N]
// starting at row `rowOff` (row stride ldin). Output row n holds column n of
// the source, contiguous in K -> B fragments become contiguous b128 loads.
// ---------------------------------------------------------------------------
__global__ void transpose_pad_kernel(const float* __restrict__ in, int ldin, int rowOff,
                                     __bf16* __restrict__ out, int N, int K, int K32,
                                     long long total)
{
  long long id = (long long)blockIdx.x * blockDim.x + threadIdx.x;
  if (id >= total) return;
  int k = (int)(id % K32);
  long long n = id / K32;
  float v = (n < N && k < K) ? in[(size_t)(rowOff + k) * ldin + n] : 0.0f;
  out[id] = (__bf16)v;
}

// ---------------------------------------------------------------------------
// Fragment load helpers (CDNA5 ISA 7.12.2 wave32 layouts, contiguous b128)
// ---------------------------------------------------------------------------
__device__ __forceinline__ v16bf load_afrag(const __bf16* aBase, int k0) {
  v8bf a0 = *(const v8bf*)(aBase + k0);         // K = half*8 + {0..7}
  v8bf a1 = *(const v8bf*)(aBase + k0 + 16);    // K = half*8 + 16 + {0..7}
  return __builtin_shufflevector(a0, a1, 0,1,2,3,4,5,6,7,8,9,10,11,12,13,14,15);
}
__device__ __forceinline__ v16bf load_bfrag(const __bf16* bBase, int k0) {
  v8bf b0 = *(const v8bf*)(bBase + k0);         // K = half*8 + {0..7}
  v8bf b1 = *(const v8bf*)(bBase + k0 + 16);    // K = half*8 + 16 + {0..7}
  return __builtin_shufflevector(b0, b1, 0,1,2,3,4,5,6,7,8,9,10,11,12,13,14,15);
}

// ---------------------------------------------------------------------------
// Wave-level bf16 WMMA GEMM, one 32x32 C macro-tile per wave (2x2 WMMA tiles,
// 4 accumulators): 8 b128 loads + 4 WMMAs per K-chunk -> 16 FLOP/byte reuse.
//   C = A(bf16, MxK, row-major, lda) * Bt^T  [+bias] [+Cin]
// Bt is the transposed-padded weight (ceil32(N) x K32). M % 32 == 0.
// ---------------------------------------------------------------------------
__global__ void gemm_bf16_kernel(const __bf16* __restrict__ A, int lda,
                                 const __bf16* __restrict__ Bt, int K32,
                                 const float* __restrict__ bias,
                                 const float* __restrict__ Cin, int ldcin,
                                 float* __restrict__ Cf, __bf16* __restrict__ Cb,
                                 int ldc, int ntiles, int N, long long tiles)
{
  long long wave = (long long)blockIdx.x * (blockDim.x >> 5) + (threadIdx.x >> 5);
  if (wave >= tiles) return;
  int mt = (int)(wave / ntiles);
  int nt = (int)(wave - (long long)mt * ntiles);

  int lane = threadIdx.x & 31;
  int half = lane >> 4;              // 0: lanes 0-15, 1: lanes 16-31
  int l16  = lane & 15;

  int arow0 = mt * 32 + l16;
  int bcol0 = nt * 32 + l16;
  const __bf16* aB0 = A  + (size_t)arow0 * lda + half * 8;
  const __bf16* aB1 = aB0 + (size_t)16 * lda;
  const __bf16* bB0 = Bt + (size_t)bcol0 * K32 + half * 8;
  const __bf16* bB1 = bB0 + (size_t)16 * K32;

  v8f acc00 = {0.f,0.f,0.f,0.f,0.f,0.f,0.f,0.f};
  v8f acc01 = acc00, acc10 = acc00, acc11 = acc00;

  for (int k0 = 0; k0 < K32; k0 += 32) {
    v16bf a0 = load_afrag(aB0, k0);
    v16bf a1 = load_afrag(aB1, k0);
    v16bf b0 = load_bfrag(bB0, k0);
    v16bf b1 = load_bfrag(bB1, k0);
    acc00 = __builtin_amdgcn_wmma_f32_16x16x32_bf16(false, a0, false, b0, (short)0, acc00, false, false);
    acc01 = __builtin_amdgcn_wmma_f32_16x16x32_bf16(false, a0, false, b1, (short)0, acc01, false, false);
    acc10 = __builtin_amdgcn_wmma_f32_16x16x32_bf16(false, a1, false, b0, (short)0, acc10, false, false);
    acc11 = __builtin_amdgcn_wmma_f32_16x16x32_bf16(false, a1, false, b1, (short)0, acc11, false, false);
  }

  // C/D layout: lane col = tile_col + l16, vgpr r -> row = tile_row + 8*half + r
#pragma unroll
  for (int cs = 0; cs < 2; ++cs) {
    int bcol = nt * 32 + cs * 16 + l16;
    if (bcol >= N) continue;
    float bv = bias ? bias[bcol] : 0.0f;
    const v8f* aLo = cs ? &acc01 : &acc00;   // rows mt*32 + 0..15
    const v8f* aHi = cs ? &acc11 : &acc10;   // rows mt*32 + 16..31
#pragma unroll
    for (int r = 0; r < 8; ++r) {
      int row0 = mt * 32 + half * 8 + r;
      int row1 = row0 + 16;
      float v0 = (*aLo)[r] + bv;
      float v1 = (*aHi)[r] + bv;
      if (Cin) {
        v0 += Cin[(size_t)row0 * ldcin + bcol];
        v1 += Cin[(size_t)row1 * ldcin + bcol];
      }
      if (Cf) {
        Cf[(size_t)row0 * ldc + bcol] = v0;
        Cf[(size_t)row1 * ldc + bcol] = v1;
      }
      if (Cb) {
        Cb[(size_t)row0 * ldc + bcol] = (__bf16)v0;
        Cb[(size_t)row1 * ldc + bcol] = (__bf16)v1;
      }
    }
  }
}

// ---------------------------------------------------------------------------
// Fused embedding gather + projection GEMM (16x16 tiles):
//   C[(t*32+b), :] = bf16(W_emb[idx[b*L+t], :]) @ embProj   (K = EMB = 300)
// Fast b128 path for full chunks; guarded tail so we never read past W_emb.
// ---------------------------------------------------------------------------
__global__ void gemm_embed_kernel(const float* __restrict__ Wemb,
                                  const int* __restrict__ idx, int L,
                                  const __bf16* __restrict__ Bt, int K32, int K,
                                  __bf16* __restrict__ Cb, int ldc,
                                  int ntiles, int N, long long tiles)
{
  long long wave = (long long)blockIdx.x * (blockDim.x >> 5) + (threadIdx.x >> 5);
  if (wave >= tiles) return;
  int mt = (int)(wave / ntiles);
  int nt = (int)(wave - (long long)mt * ntiles);

  int lane = threadIdx.x & 31;
  int half = lane >> 4;
  int l16  = lane & 15;
  int arow = mt * 16 + l16;
  int bcol = nt * 16 + l16;

  int brow = arow & 31;            // batch
  int tpos = arow >> 5;            // time
  int word = idx[(size_t)brow * L + tpos];
  const float*  arp   = Wemb + (size_t)word * K + half * 8;
  const __bf16* bBase = Bt + (size_t)bcol * K32 + half * 8;

  v8f acc = {0.f, 0.f, 0.f, 0.f, 0.f, 0.f, 0.f, 0.f};

  for (int k0 = 0; k0 < K32; k0 += 32) {
    v16bf av;
    if (k0 + 32 <= K) {
      v4f f0 = *(const v4f*)(arp + k0);
      v4f f1 = *(const v4f*)(arp + k0 + 4);
      v4f f2 = *(const v4f*)(arp + k0 + 16);
      v4f f3 = *(const v4f*)(arp + k0 + 20);
      v8bf a0 = __builtin_shufflevector(__builtin_convertvector(f0, v4bf),
                                        __builtin_convertvector(f1, v4bf),
                                        0,1,2,3,4,5,6,7);
      v8bf a1 = __builtin_shufflevector(__builtin_convertvector(f2, v4bf),
                                        __builtin_convertvector(f3, v4bf),
                                        0,1,2,3,4,5,6,7);
      av = __builtin_shufflevector(a0, a1, 0,1,2,3,4,5,6,7,8,9,10,11,12,13,14,15);
    } else {
#pragma unroll
      for (int e = 0; e < 16; ++e) {
        int ka = k0 + half * 8 + (e < 8 ? e : e + 8);
        av[e] = (ka < K) ? (__bf16)Wemb[(size_t)word * K + ka] : (__bf16)0.0f;
      }
    }
    v16bf bv = load_bfrag(bBase, k0);
    acc = __builtin_amdgcn_wmma_f32_16x16x32_bf16(false, av, false, bv,
                                                  (short)0, acc, false, false);
  }

  if (bcol < N) {
#pragma unroll
    for (int r = 0; r < 8; ++r) {
      int row = mt * 16 + half * 8 + r;
      Cb[(size_t)row * ldc + bcol] = (__bf16)acc[r];
    }
  }
}

// ---------------------------------------------------------------------------
// Elementwise / reduction kernels
// ---------------------------------------------------------------------------
__device__ __forceinline__ float sigmoidf_(float x) { return 1.f / (1.f + __expf(-x)); }

// g: (32 x 2400) pre-activation gates [i|f|g|o]; c updated in place.
__global__ void lstm_cell_kernel(const float* __restrict__ g,
                                 float* __restrict__ c,
                                 __bf16* __restrict__ hBf,
                                 float* __restrict__ hF)
{
  int id = blockIdx.x * blockDim.x + threadIdx.x;
  if (id >= kB * kH) return;
  int b = id / kH, j = id - b * kH;
  const float* gr = g + (size_t)b * kH4;
  float si = sigmoidf_(gr[j]);
  float sf = sigmoidf_(gr[kH + j]);
  float gg = tanhf(gr[2 * kH + j]);
  float so = sigmoidf_(gr[3 * kH + j]);
  float cn = sf * c[id] + si * gg;
  c[id] = cn;
  float h = so * tanhf(cn);
  hBf[id] = (__bf16)h;
  if (hF) hF[id] = h;
}

__global__ void concat_bf_kernel(const __bf16* __restrict__ X, const __bf16* __restrict__ Y,
                                 __bf16* __restrict__ out, int n, int w) {
  int id = blockIdx.x * blockDim.x + threadIdx.x;
  if (id >= n) return;
  int r = id / (2 * w), d = id - r * 2 * w;
  out[id] = (d < w) ? X[(size_t)r * w + d] : Y[(size_t)r * w + (d - w)];
}

__global__ void concat_f2bf_kernel(const float* __restrict__ X, const float* __restrict__ Y,
                                   __bf16* __restrict__ out, int n, int w) {
  int id = blockIdx.x * blockDim.x + threadIdx.x;
  if (id >= n) return;
  int r = id / (2 * w), d = id - r * 2 * w;
  out[id] = (__bf16)((d < w) ? X[(size_t)r * w + d] : Y[(size_t)r * w + (d - w)]);
}

// enc[b,c,:] = [hsF[c,b,:], hsB[c,b,:]]  (bf16)
__global__ void build_enc_kernel(const __bf16* __restrict__ hsF,
                                 const __bf16* __restrict__ hsB,
                                 __bf16* __restrict__ enc) {
  int id = blockIdx.x * blockDim.x + threadIdx.x;
  if (id >= kB * kCL * kH2) return;
  int d  = id % kH2;
  int bc = id / kH2;
  int c  = bc % kCL;
  int b  = bc / kCL;
  enc[id] = (d < kH) ? hsF[((size_t)c * kB + b) * kH + d]
                     : hsB[((size_t)c * kB + b) * kH + (d - kH)];
}

// e[b,c] = hd[b,:] . encProj[b*CL+c,:]  masked by cw!=0
__global__ void attn_scores_kernel(const float* __restrict__ hd,
                                   const float* __restrict__ encProj,
                                   const int* __restrict__ cw,
                                   float* __restrict__ e) {
  int id = blockIdx.x * blockDim.x + threadIdx.x;
  if (id >= kB * kCL) return;
  int b = id / kCL, c = id - b * kCL;
  const float* hp = hd + (size_t)b * kH;
  const float* ep = encProj + (size_t)id * kH;
  float s = 0.f;
  for (int j = 0; j < kH; ++j) s += hp[j] * ep[j];
  e[id] = (cw[(size_t)b * kCL + c] != 0) ? s : -INFINITY;
}

// in-place softmax over rows of length n (block per row)
__global__ void softmax_rows_kernel(float* __restrict__ e, int n) {
  __shared__ float sh[256];
  float* row = e + (size_t)blockIdx.x * n;
  float m = -INFINITY;
  for (int i = threadIdx.x; i < n; i += blockDim.x) m = fmaxf(m, row[i]);
  sh[threadIdx.x] = m; __syncthreads();
  for (int o = blockDim.x >> 1; o > 0; o >>= 1) {
    if (threadIdx.x < o) sh[threadIdx.x] = fmaxf(sh[threadIdx.x], sh[threadIdx.x + o]);
    __syncthreads();
  }
  m = sh[0]; __syncthreads();
  float s = 0.f;
  for (int i = threadIdx.x; i < n; i += blockDim.x) s += __expf(row[i] - m);
  sh[threadIdx.x] = s; __syncthreads();
  for (int o = blockDim.x >> 1; o > 0; o >>= 1) {
    if (threadIdx.x < o) sh[threadIdx.x] += sh[threadIdx.x + o];
    __syncthreads();
  }
  float inv = 1.f / sh[0];
  for (int i = threadIdx.x; i < n; i += blockDim.x) row[i] = __expf(row[i] - m) * inv;
}

// a[b,d] = sum_c alpha[b,c] * enc[b,c,d]  -> bf16
__global__ void attn_apply_kernel(const float* __restrict__ alpha,
                                  const __bf16* __restrict__ enc,
                                  __bf16* __restrict__ aOut) {
  int id = blockIdx.x * blockDim.x + threadIdx.x;
  if (id >= kB * kH2) return;
  int b = id / kH2, d = id - b * kH2;
  const float*  al = alpha + (size_t)b * kCL;
  const __bf16* ep = enc + (size_t)b * kCL * kH2 + d;
  float s = 0.f;
  for (int c = 0; c < kCL; ++c) s += al[c] * (float)ep[(size_t)c * kH2];
  aOut[id] = (__bf16)s;
}

// O_t = tanh(tmp); store at Xout[(b*QL + t), :] (row layout = output layout)
__global__ void tanh_store_kernel(const float* __restrict__ t, __bf16* __restrict__ Xout, int tstep) {
  int id = blockIdx.x * blockDim.x + threadIdx.x;
  if (id >= kB * kH) return;
  int b = id / kH, j = id - b * kH;
  Xout[((size_t)b * kQL + tstep) * kH + j] = (__bf16)tanhf(t[id]);
}

__global__ void log_softmax_kernel(float* __restrict__ x, int n) {
  __shared__ float sh[512];
  float* row = x + (size_t)blockIdx.x * n;
  float m = -INFINITY;
  for (int i = threadIdx.x; i < n; i += blockDim.x) m = fmaxf(m, row[i]);
  sh[threadIdx.x] = m; __syncthreads();
  for (int o = blockDim.x >> 1; o > 0; o >>= 1) {
    if (threadIdx.x < o) sh[threadIdx.x] = fmaxf(sh[threadIdx.x], sh[threadIdx.x + o]);
    __syncthreads();
  }
  m = sh[0]; __syncthreads();
  float s = 0.f;
  for (int i = threadIdx.x; i < n; i += blockDim.x) s += __expf(row[i] - m);
  sh[threadIdx.x] = s; __syncthreads();
  for (int o = blockDim.x >> 1; o > 0; o >>= 1) {
    if (threadIdx.x < o) sh[threadIdx.x] += sh[threadIdx.x + o];
    __syncthreads();
  }
  float lse = m + logf(sh[0]);
  for (int i = threadIdx.x; i < n; i += blockDim.x) row[i] -= lse;
}

// ---------------------------------------------------------------------------
// Host-side launch helpers
// ---------------------------------------------------------------------------
static inline int ceil32i(int x) { return (x + 31) & ~31; }

static void make_tw(hipStream_t s, const float* in, int ldin, int rowOff,
                    __bf16* out, int N, int K)
{
  int K32 = ceil32i(K), N32 = ceil32i(N);
  long long total = (long long)N32 * K32;
  transpose_pad_kernel<<<(int)((total + 255) / 256), 256, 0, s>>>(
      in, ldin, rowOff, out, N, K, K32, total);
}

static void gemm(hipStream_t s, const __bf16* A, int lda, const __bf16* Bt, int K32,
                 const float* bias, const float* Cin, int ldcin,
                 float* Cf, __bf16* Cb, int ldc, int M, int N)
{
  int mt = M / 32, nt = (N + 31) / 32;                 // M is a multiple of 32
  long long tiles = (long long)mt * nt;
  int blocks = (int)((tiles + 3) / 4);                 // 4 waves (128 thr) / block
  gemm_bf16_kernel<<<blocks, 128, 0, s>>>(A, lda, Bt, K32, bias, Cin, ldcin,
                                          Cf, Cb, ldc, nt, N, tiles);
}

static void gemm_embed(hipStream_t s, const float* Wemb, const int* idx, int L,
                       const __bf16* Bt, int K32, int K,
                       __bf16* Cb, int ldc, int M, int N)
{
  int mt = M / 16, nt = (N + 15) / 16;
  long long tiles = (long long)mt * nt;
  int blocks = (int)((tiles + 3) / 4);
  gemm_embed_kernel<<<blocks, 128, 0, s>>>(Wemb, idx, L, Bt, K32, K, Cb, ldc,
                                           nt, N, tiles);
}

// ---------------------------------------------------------------------------
extern "C" void kernel_launch(void* const* d_in, const int* in_sizes, int n_in,
                              void* d_out, int out_size, void* d_ws, size_t ws_size,
                              hipStream_t stream)
{
  (void)in_sizes; (void)n_in; (void)out_size; (void)ws_size;

  const int*   cw       = (const int*)  d_in[0];
  const int*   qw       = (const int*)  d_in[1];
  const float* W_emb    = (const float*)d_in[2];
  const float* emb_proj = (const float*)d_in[3];
  const float* Wx_f     = (const float*)d_in[4];
  const float* Wh_f     = (const float*)d_in[5];
  const float* b_f      = (const float*)d_in[6];
  const float* Wx_b     = (const float*)d_in[7];
  const float* Wh_b     = (const float*)d_in[8];
  const float* b_b      = (const float*)d_in[9];
  const float* Wh_init  = (const float*)d_in[10];
  const float* Wc_init  = (const float*)d_in[11];
  const float* Wx_d     = (const float*)d_in[12];
  const float* Wh_d     = (const float*)d_in[13];
  const float* b_d      = (const float*)d_in[14];
  const float* W_att    = (const float*)d_in[15];
  const float* W_comb   = (const float*)d_in[16];
  const float* W_gen    = (const float*)d_in[17];
  const float* b_gen    = (const float*)d_in[18];
  float* out = (float*)d_out;

  // padded dims for transposed weights
  const int K32_H   = ceil32i(kH);    // 608
  const int K32_H2  = ceil32i(kH2);   // 1216
  const int K32_EMB = ceil32i(kEMB);  // 320
  const int N32_H   = ceil32i(kH);    // 608
  const int N32_V   = ceil32i(kV);    // 50016

  // ---- workspace carving ----
  char* ws = (char*)d_ws;
  size_t off = 0;
  auto carve = [&](size_t bytes) -> void* {
    void* p = ws + off;
    off = (off + bytes + 255) & ~(size_t)255;
    return p;
  };
  auto cb16 = [&](size_t n) -> __bf16* { return (__bf16*)carve(n * 2); };
  auto cf32 = [&](size_t n) -> float*  { return (float*) carve(n * 4); };

  // transposed + padded bf16 weights (ceil32(N) x K32)
  __bf16* embProjT = cb16((size_t)N32_H * K32_EMB);
  __bf16* WxfT     = cb16((size_t)kH4 * K32_H);
  __bf16* WhfT     = cb16((size_t)kH4 * K32_H);
  __bf16* WxbT     = cb16((size_t)kH4 * K32_H);
  __bf16* WhbT     = cb16((size_t)kH4 * K32_H);
  __bf16* WhInitT  = cb16((size_t)N32_H * K32_H2);
  __bf16* WcInitT  = cb16((size_t)N32_H * K32_H2);
  __bf16* WxdTq    = cb16((size_t)kH4 * K32_H);   // Wx_d rows [0,600)
  __bf16* WxdTo    = cb16((size_t)kH4 * K32_H);   // Wx_d rows [600,1200)
  __bf16* WhdT     = cb16((size_t)kH4 * K32_H);
  __bf16* WattT    = cb16((size_t)N32_H * K32_H2);
  __bf16* WcombTh  = cb16((size_t)N32_H * K32_H); // W_comb rows [0,600)
  __bf16* WcombTa  = cb16((size_t)N32_H * K32_H2);// W_comb rows [600,1800)
  __bf16* WgenT    = cb16((size_t)N32_V * K32_H);
  // bf16 activations
  __bf16* Xc   = cb16((size_t)kCL * kB * kH);   // rows (t*32+b)
  __bf16* Xq   = cb16((size_t)kQL * kB * kH);   // rows (t*32+b)
  __bf16* hsF  = cb16((size_t)kCL * kB * kH);
  __bf16* hsB  = cb16((size_t)kCL * kB * kH);
  __bf16* enc  = cb16((size_t)kB * kCL * kH2);
  __bf16* hcat = cb16((size_t)kB * kH2);
  __bf16* ccat = cb16((size_t)kB * kH2);
  __bf16* hdBf = cb16((size_t)kB * kH);
  __bf16* aBf  = cb16((size_t)kB * kH2);
  __bf16* Xout = cb16((size_t)kB * kQL * kH);   // rows (b*QL+t) == output order
  __bf16* zero = cb16((size_t)kB * kH);
  // fp32 scratch
  float* Gf      = cf32((size_t)kCL * kB * kH4);
  float* Gb      = cf32((size_t)kCL * kB * kH4);
  float* encProj = cf32((size_t)kB * kCL * kH);
  float* gBuf    = cf32((size_t)kB * kH4);
  float* cF      = cf32((size_t)kB * kH);
  float* cB      = cf32((size_t)kB * kH);
  float* hdF     = cf32((size_t)kB * kH);
  float* cD      = cf32((size_t)kB * kH);
  float* eBuf    = cf32((size_t)kB * kCL);
  float* tmpO    = cf32((size_t)kB * kH);
  (void)carve(256);                             // tail slack for K32 over-reads

  // ---- per-call state init (deterministic; graph-capturable) ----
  hipMemsetAsync(zero, 0, (size_t)kB * kH * 2, stream);
  hipMemsetAsync(cF,   0, (size_t)kB * kH * 4, stream);
  hipMemsetAsync(cB,   0, (size_t)kB * kH * 4, stream);

  // ---- weight prep: fp32 -> transposed, zero-padded bf16 ----
  make_tw(stream, emb_proj, kH,  0,   embProjT, kH,  kEMB);
  make_tw(stream, Wx_f,    kH4,  0,   WxfT,     kH4, kH);
  make_tw(stream, Wh_f,    kH4,  0,   WhfT,     kH4, kH);
  make_tw(stream, Wx_b,    kH4,  0,   WxbT,     kH4, kH);
  make_tw(stream, Wh_b,    kH4,  0,   WhbT,     kH4, kH);
  make_tw(stream, Wh_init, kH,   0,   WhInitT,  kH,  kH2);
  make_tw(stream, Wc_init, kH,   0,   WcInitT,  kH,  kH2);
  make_tw(stream, Wx_d,    kH4,  0,   WxdTq,    kH4, kH);
  make_tw(stream, Wx_d,    kH4,  kH,  WxdTo,    kH4, kH);
  make_tw(stream, Wh_d,    kH4,  0,   WhdT,     kH4, kH);
  make_tw(stream, W_att,   kH,   0,   WattT,    kH,  kH2);
  make_tw(stream, W_comb,  kH,   0,   WcombTh,  kH,  kH);
  make_tw(stream, W_comb,  kH,   kH,  WcombTa,  kH,  kH2);
  make_tw(stream, W_gen,   kV,   0,   WgenT,    kV,  kH);

  // ---- embeddings: X = bf16(W_emb[idx]) @ emb_proj ----
  gemm_embed(stream, W_emb, cw, kCL, embProjT, K32_EMB, kEMB, Xc, kH, kCL * kB, kH);
  gemm_embed(stream, W_emb, qw, kQL, embProjT, K32_EMB, kEMB, Xq, kH, kQL * kB, kH);

  // ---- time-batched input gates: G = X @ Wx + b  (big WMMA GEMMs) ----
  gemm(stream, Xc, kH, WxfT, K32_H, b_f, nullptr, 0, Gf, nullptr, kH4, kCL * kB, kH4);
  gemm(stream, Xc, kH, WxbT, K32_H, b_b, nullptr, 0, Gb, nullptr, kH4, kCL * kB, kH4);

  // ---- forward LSTM (sequential over 400 steps) ----
  for (int t = 0; t < kCL; ++t) {
    const __bf16* hprev = (t == 0) ? zero : hsF + (size_t)(t - 1) * kB * kH;
    gemm(stream, hprev, kH, WhfT, K32_H, nullptr,
         Gf + (size_t)t * kB * kH4, kH4, gBuf, nullptr, kH4, kB, kH4);
    lstm_cell_kernel<<<(kB * kH + 255) / 256, 256, 0, stream>>>(
        gBuf, cF, hsF + (size_t)t * kB * kH, nullptr);
  }
  // ---- backward LSTM ----
  for (int s = 0; s < kCL; ++s) {
    int t = kCL - 1 - s;
    const __bf16* hprev = (s == 0) ? zero : hsB + (size_t)(t + 1) * kB * kH;
    gemm(stream, hprev, kH, WhbT, K32_H, nullptr,
         Gb + (size_t)t * kB * kH4, kH4, gBuf, nullptr, kH4, kB, kH4);
    lstm_cell_kernel<<<(kB * kH + 255) / 256, 256, 0, stream>>>(
        gBuf, cB, hsB + (size_t)t * kB * kH, nullptr);
  }

  // ---- decoder init: h0 = [hf|hb] @ Wh_init, c0 = [cf|cb] @ Wc_init ----
  concat_bf_kernel<<<(kB * kH2 + 255) / 256, 256, 0, stream>>>(
      hsF + (size_t)(kCL - 1) * kB * kH, hsB /* t=0 */, hcat, kB * kH2, kH);
  concat_f2bf_kernel<<<(kB * kH2 + 255) / 256, 256, 0, stream>>>(
      cF, cB, ccat, kB * kH2, kH);
  gemm(stream, hcat, kH2, WhInitT, K32_H2, nullptr, nullptr, 0, hdF, hdBf, kH, kB, kH);
  gemm(stream, ccat, kH2, WcInitT, K32_H2, nullptr, nullptr, 0, cD, nullptr, kH, kB, kH);

  // ---- enc = concat(hs_f, hs_b) and enc_proj = enc @ W_att ----
  build_enc_kernel<<<(kB * kCL * kH2 + 255) / 256, 256, 0, stream>>>(hsF, hsB, enc);
  gemm(stream, enc, kH2, WattT, K32_H2, nullptr, nullptr, 0, encProj, nullptr, kH,
       kB * kCL, kH);

  // ---- attention decoder: 30 steps ----
  for (int t = 0; t < kQL; ++t) {
    // g = hd @ Wh_d + b_d
    gemm(stream, hdBf, kH, WhdT, K32_H, b_d, nullptr, 0, gBuf, nullptr, kH4, kB, kH4);
    // g += q_t @ Wx_d[0:600]
    gemm(stream, Xq + (size_t)t * kB * kH, kH, WxdTq, K32_H, nullptr,
         gBuf, kH4, gBuf, nullptr, kH4, kB, kH4);
    // g += O_prev @ Wx_d[600:1200]
    const __bf16* Oprev = (t == 0) ? zero : Xout + (size_t)(t - 1) * kH;
    int ldOprev = (t == 0) ? kH : kQL * kH;
    gemm(stream, Oprev, ldOprev, WxdTo, K32_H, nullptr,
         gBuf, kH4, gBuf, nullptr, kH4, kB, kH4);
    // cell update -> hd (f32 + bf16), cD in place
    lstm_cell_kernel<<<(kB * kH + 255) / 256, 256, 0, stream>>>(gBuf, cD, hdBf, hdF);
    // e = hd . enc_proj + mask ; softmax ; a = alpha @ enc
    attn_scores_kernel<<<(kB * kCL + 255) / 256, 256, 0, stream>>>(hdF, encProj, cw, eBuf);
    softmax_rows_kernel<<<kB, 256, 0, stream>>>(eBuf, kCL);
    attn_apply_kernel<<<(kB * kH2 + 255) / 256, 256, 0, stream>>>(eBuf, enc, aBf);
    // O = tanh([hd|a] @ W_comb)
    gemm(stream, hdBf, kH, WcombTh, K32_H, nullptr, nullptr, 0, tmpO, nullptr, kH, kB, kH);
    gemm(stream, aBf, kH2, WcombTa, K32_H2, nullptr,
         tmpO, kH, tmpO, nullptr, kH, kB, kH);
    tanh_store_kernel<<<(kB * kH + 255) / 256, 256, 0, stream>>>(tmpO, Xout, t);
  }

  // ---- vocab projection + log_softmax (rows already in (b,t) output order) ----
  gemm(stream, Xout, kH, WgenT, K32_H, b_gen, nullptr, 0, out, nullptr, kV,
       kB * kQL, kV);
  log_softmax_kernel<<<kB * kQL, 512, 0, stream>>>(out, kV);
}
